// NdeviceTopModel_738734375410
// MI455X (gfx1250) — matrix operand order repaired and checked by reading, not compiled, not run on hardware
//
#include <hip/hip_runtime.h>
#include <hip/hip_bf16.h>

typedef __attribute__((ext_vector_type(16))) __bf16       v16bf;
typedef __attribute__((ext_vector_type(8)))  float        v8f;
typedef __attribute__((ext_vector_type(4)))  unsigned int uint4v;

#define B_    512      // batch
#define D_    512      // feature dim (K)
#define NTILE 128      // columns per block
#define PITCH 520      // bf16 elems per LDS row (512 + 8 pad)

__device__ __forceinline__ unsigned short f2bf(float f) {
    unsigned int u = __float_as_uint(f);
    unsigned int r = u + 0x7FFFu + ((u >> 16) & 1u);   // round-to-nearest-even
    return (unsigned short)(r >> 16);
}
__device__ __forceinline__ float bf2f(unsigned short h) {
    return __uint_as_float(((unsigned int)h) << 16);
}

union FragBF { v16bf v; uint4v q[2]; };

// ---------------- Kernel 1: row-normalize features, emit bf16 ----------------
__global__ void k_feat_normalize(const float* __restrict__ x,
                                 unsigned short* __restrict__ A) {
    __shared__ float red[256];
    const int m = blockIdx.x, t = threadIdx.x;
    float x0 = x[m * D_ + t];
    float x1 = x[m * D_ + t + 256];
    red[t] = x0 * x0 + x1 * x1;
    __syncthreads();
    for (int s = 128; s > 0; s >>= 1) {
        if (t < s) red[t] += red[t + s];
        __syncthreads();
    }
    float inv = 1.0f / fmaxf(sqrtf(red[0]), 1e-12f);
    A[m * D_ + t]       = f2bf(x0 * inv);
    A[m * D_ + t + 256] = f2bf(x1 * inv);
}

// ---------------- Kernel 2: zero accumulators ----------------
__global__ void k_init(float* __restrict__ S, float* __restrict__ T) {
    int t = threadIdx.x;
    S[t] = 0.f; S[t + 256] = 0.f;
    T[t] = 0.f; T[t + 256] = 0.f;
}

// ---------------- epilogue helper: one 16x16 f32 tile ----------------
__device__ __forceinline__ void tile_epilogue(const v8f& acc, int m0, int hi,
                                              int l15, int gcol, float inv,
                                              const int* __restrict__ lab,
                                              float* __restrict__ Sl,
                                              float* __restrict__ T) {
#pragma unroll
    for (int r = 0; r < 8; ++r) {
        int   m = m0 + r + hi * 8;
        float v = fminf(fmaxf(acc[r] * inv, -1.f), 1.f);   // clip cosine
        float logit = 64.f * v;
        if (gcol == lab[m]) {                               // CosFace margin
            logit -= 22.4f;                                 // 64 * 0.35
            T[m] = logit;                                   // true-label logit
        }
        float e = __expf(logit - 64.f);                     // fixed shift (logit<=64)
        e += __shfl_xor(e, 1, 16);                          // reduce over 16 cols
        e += __shfl_xor(e, 2, 16);
        e += __shfl_xor(e, 4, 16);
        e += __shfl_xor(e, 8, 16);
        if (l15 == 0) atomicAdd(&Sl[m], e);                 // ds_add_f32
    }
}

// ---------------- Kernel 3: fused normalize(w) + GEMM + margin + partial LSE ----------------
__global__ void __launch_bounds__(256)
k_cosface(const unsigned short* __restrict__ A,   // normed feat bf16 [512][512]
          const float*          __restrict__ W,   // raw weights f32 [512][C]
          const long long*      __restrict__ label,
          float* __restrict__ S, float* __restrict__ T, int C) {
    extern __shared__ unsigned char smem[];
    unsigned short* ldsw = (unsigned short*)smem;                               // 128*520*2
    float* Sl   = (float*)(smem + NTILE * PITCH * 2);                           // 512 f32
    float* invn = (float*)(smem + NTILE * PITCH * 2 + 512 * 4);                 // 128 f32
    int*   lab  = (int*)  (smem + NTILE * PITCH * 2 + 512 * 4 + 128 * 4);       // 512 i32

    const int tid   = threadIdx.x;
    const int cbase = blockIdx.x * NTILE;

    for (int i = tid; i < B_; i += 256) { Sl[i] = 0.f; lab[i] = (int)label[i]; }

    // Stage W tile -> LDS, transposed [n][k], bf16 (w read from HBM exactly once)
    for (int i = tid; i < NTILE * D_; i += 256) {
        int k = i >> 7;            // 0..511
        int n = i & (NTILE - 1);   // 0..127, consecutive tid -> consecutive c (coalesced)
        int c = cbase + n;
        float v = (c < C) ? W[(long long)k * C + c] : 0.f;
        ldsw[n * PITCH + k] = f2bf(v);
    }
    __syncthreads();

    // Per-column inverse norms from the staged tile (no extra HBM pass)
    if (tid < NTILE) {
        float s = 0.f;
        const unsigned short* row = ldsw + tid * PITCH;
        for (int k = 0; k < D_; ++k) { float v = bf2f(row[k]); s += v * v; }
        invn[tid] = 1.0f / fmaxf(sqrtf(s), 1e-12f);
    }
    __syncthreads();

    const int lane = tid & 31;
    const int wid  = tid >> 5;
    const int l15  = lane & 15;
    const int hi   = lane >> 4;
    const int nloc = wid * 16 + l15;        // wave owns 16 columns
    const int gcol = cbase + nloc;
    const float inv = invn[nloc];

    const __bf16* bbase = (const __bf16*)ldsw + nloc * PITCH + hi * 8;
    const __bf16* Ab    = (const __bf16*)A;

    for (int mo = 0; mo < B_ / 32; ++mo) {   // 32 batch rows per iteration
        const int m0 = mo * 32;
        v8f acc0{}; v8f acc1{};
        const __bf16* a0p = Ab + (m0 + l15) * D_ + hi * 8;
        const __bf16* a1p = a0p + 16 * D_;
#pragma unroll 4
        for (int kk = 0; kk < D_ / 32; ++kk) {
            const int k = kk * 32;
            FragBF bf, af0, af1;
            // B: per ISA layout, lo lanes K{0-7,16-23}, hi lanes K{8-15,24-31}
            bf.q[0] = *(const uint4v*)(bbase + k);
            bf.q[1] = *(const uint4v*)(bbase + k + 16);
            // A: per ISA layout, lo lanes K{0-7,16-23}, hi lanes K{8-15,24-31}
            af0.q[0] = *(const uint4v*)(a0p + k);
            af0.q[1] = *(const uint4v*)(a0p + k + 16);
            af1.q[0] = *(const uint4v*)(a1p + k);
            af1.q[1] = *(const uint4v*)(a1p + k + 16);
            acc0 = __builtin_amdgcn_wmma_f32_16x16x32_bf16(
                       false, af0.v, false, bf.v, (short)0, acc0, false, false);
            acc1 = __builtin_amdgcn_wmma_f32_16x16x32_bf16(
                       false, af1.v, false, bf.v, (short)0, acc1, false, false);
        }
        tile_epilogue(acc0, m0,      hi, l15, gcol, inv, lab, Sl, T);
        tile_epilogue(acc1, m0 + 16, hi, l15, gcol, inv, lab, Sl, T);
    }

    __syncthreads();
    for (int i = tid; i < B_; i += 256) atomicAdd(&S[i], Sl[i]);
}

// ---------------- Kernel 4: final reduction to scalar loss ----------------
__global__ void k_finalize(const float* __restrict__ S,
                           const float* __restrict__ T,
                           float* __restrict__ out) {
    __shared__ float red[256];
    int t = threadIdx.x;
    float a = (logf(S[t])       + 64.f - T[t]) +
              (logf(S[t + 256]) + 64.f - T[t + 256]);
    red[t] = a;
    __syncthreads();
    for (int s = 128; s > 0; s >>= 1) {
        if (t < s) red[t] += red[t + s];
        __syncthreads();
    }
    if (t == 0) out[0] = red[0] / (float)B_;
}

extern "C" void kernel_launch(void* const* d_in, const int* in_sizes, int n_in,
                              void* d_out, int out_size, void* d_ws, size_t ws_size,
                              hipStream_t stream) {
    const float*     x   = (const float*)d_in[0];      // [512,512] f32
    const long long* lbl = (const long long*)d_in[1];  // [512] i64
    const float*     w   = (const float*)d_in[2];      // [512,C] f32
    float*           out = (float*)d_out;

    const int Bm = in_sizes[1];          // 512
    const int Dm = in_sizes[0] / Bm;     // 512
    const int C  = in_sizes[2] / Dm;     // 100000

    unsigned short* A = (unsigned short*)d_ws;                    // bf16 feat
    float* S = (float*)((char*)d_ws + (size_t)Bm * Dm * 2);       // sumexp per row
    float* T = S + Bm;                                            // true-label logit

    const size_t smem = (size_t)NTILE * PITCH * 2 + 512 * 4 + 128 * 4 + 512 * 4;
    hipFuncSetAttribute((const void*)k_cosface,
                        hipFuncAttributeMaxDynamicSharedMemorySize, (int)smem);

    k_feat_normalize<<<Bm, 256, 0, stream>>>(x, A);
    k_init<<<1, 256, 0, stream>>>(S, T);
    const int nb = (C + NTILE - 1) / NTILE;
    k_cosface<<<nb, 256, smem, stream>>>(A, w, lbl, S, T, C);
    k_finalize<<<1, 256, 0, stream>>>(S, T, out);
}
